// CapsuleNetwork_88905823027539
// MI455X (gfx1250) — compile-verified, matrix operand order checked by reading, not run on hardware
//
#include <hip/hip_runtime.h>
#include <hip/hip_bf16.h>

// ---------------------------------------------------------------------------
// CDNA5 / gfx1250 CapsuleNetwork forward.  All GEMMs run on
// v_wmma_f32_16x16x32_bf16 (wave32, 16x16 tiles, K=32).  Conv layers are
// implicit-im2col GEMMs gathering from the L2-resident bf16 activations.
// A/B tiles use GLOBAL_LOAD_ASYNC_TO_LDS (ASYNCcnt) when the toolchain
// exposes the gfx1250 async-to-LDS builtins.
// ---------------------------------------------------------------------------

typedef __attribute__((ext_vector_type(16))) __bf16 bf16x16;   // 8 VGPRs: WMMA A/B frag
typedef __attribute__((ext_vector_type(8)))  float  f32x8;     // 8 VGPRs: WMMA C/D frag
typedef __attribute__((ext_vector_type(4)))  float  f32x4;     // 16B vector ld/st

#define LSTR 48        // LDS row stride in bf16 (96B: keeps b128 alignment, skews banks)
#define EPSF 1e-7f

#if defined(__gfx1250__) && __has_builtin(__builtin_amdgcn_global_load_async_to_lds_b128) && \
    __has_builtin(__builtin_amdgcn_s_wait_asynccnt)
#define HAS_ASYNC_LDS 1
#else
#define HAS_ASYNC_LDS 0
#endif

#if HAS_ASYNC_LDS
// Builtin signature (from hipcc diagnostic): params are pointers to
// 'int __attribute__((vector_size(16)))' in AS1 (global) / AS3 (LDS).
typedef int v4i_vs __attribute__((vector_size(16)));
typedef __attribute__((address_space(1))) v4i_vs as1_v4i;
typedef __attribute__((address_space(3))) v4i_vs as3_v4i;
#endif

__device__ __forceinline__ __bf16 f2bf(float f) {
  unsigned u = __float_as_uint(f);
  unsigned r = (u + 0x7FFFu + ((u >> 16) & 1u)) >> 16;   // round-nearest-even
  unsigned short s = (unsigned short)r;
  __bf16 b; __builtin_memcpy(&b, &s, 2); return b;
}
__device__ __forceinline__ float bf2f(__bf16 b) {
  unsigned short s; __builtin_memcpy(&s, &b, 2);
  return __uint_as_float(((unsigned)s) << 16);
}

// 16B global -> LDS copy.  Async DMA (ASYNCcnt) on gfx1250 toolchains that
// have the builtin; synchronous vector copy otherwise.
__device__ __forceinline__ void copy16_g2l(const void* g, void* l) {
#if HAS_ASYNC_LDS
  __builtin_amdgcn_global_load_async_to_lds_b128(
      (as1_v4i*)(uintptr_t)g,
      (as3_v4i*)(unsigned)(uintptr_t)l, 0, 0);
#else
  *reinterpret_cast<f32x4*>(l) = *reinterpret_cast<const f32x4*>(g);
#endif
}
__device__ __forceinline__ void copy_wait() {
#if HAS_ASYNC_LDS
  __builtin_amdgcn_s_wait_asynccnt(0);
#endif
}

// Fragment load from LDS row (row-major [*, LSTR]):
// 16-bit A 16x32 layout: lanes 0-15 hold K0..7 (v0..3) and K16..23 (v4..7);
// lanes 16-31 hold K8..15 and K24..31.  kbase = laneHi ? 8 : 0.
__device__ __forceinline__ bf16x16 frag_ld(const __bf16* rowp, int kbase) {
  union { bf16x16 v; f32x4 q[2]; } f;
  f.q[0] = *reinterpret_cast<const f32x4*>(rowp + kbase);
  f.q[1] = *reinterpret_cast<const f32x4*>(rowp + kbase + 16);
  return f.v;
}

// One K=32 step over this wave's 64(M)x32(N) sub-tile: 4 m-tiles x 2 n-tiles.
__device__ __forceinline__ void mma_step(const __bf16* wA, const __bf16* wB,
                                         int lane, f32x8 (&acc)[4][2]) {
  const int ml    = lane & 15;
  const int kbase = (lane & 16) ? 8 : 0;
  bf16x16 bf[2];
#pragma unroll
  for (int nt = 0; nt < 2; ++nt) bf[nt] = frag_ld(wB + (nt * 16 + ml) * LSTR, kbase);
#pragma unroll
  for (int mt = 0; mt < 4; ++mt) {
    bf16x16 a = frag_ld(wA + (mt * 16 + ml) * LSTR, kbase);
#pragma unroll
    for (int nt = 0; nt < 2; ++nt)
      acc[mt][nt] = __builtin_amdgcn_wmma_f32_16x16x32_bf16(
          false, a, false, bf[nt], (short)0, acc[mt][nt], false, false);
  }
}

// ---------------------------------------------------------------------------
// Prep: cast weights/input to bf16 layouts, zero loss accumulators.
// ---------------------------------------------------------------------------
__global__ void k_prep(const float* __restrict__ input, const float* __restrict__ c1w,
                       const float* __restrict__ c2w, const float* __restrict__ wpred,
                       const float* __restrict__ d1w, const float* __restrict__ d2w,
                       const float* __restrict__ d3w,
                       __bf16* __restrict__ xbf, __bf16* __restrict__ a1,
                       __bf16* __restrict__ a2, __bf16* __restrict__ what,
                       __bf16* __restrict__ d1b, __bf16* __restrict__ d2b,
                       __bf16* __restrict__ d3b, float* __restrict__ accs) {
  const int g = blockIdx.x * blockDim.x + threadIdx.x;
  const int gs = gridDim.x * blockDim.x;
  for (int i = g; i < 256 * 784; i += gs) xbf[i] = f2bf(input[i]);
  for (int i = g; i < 256 * 96; i += gs) {               // conv1_w padded K 81->96
    int r = i / 96, c = i - r * 96;
    a1[i] = f2bf(c < 81 ? c1w[r * 81 + c] : 0.f);
  }
  for (int i = g; i < 256 * 20736; i += gs) a2[i] = f2bf(c2w[i]);
  for (int i = g; i < 1152 * 1280; i += gs) {            // W_pred[p,j,d,k] -> [p][k][j*16+d]
    int p = i / 1280, r = i - p * 1280, k = r / 160, n = r - k * 160;
    int j = n >> 4, d = n & 15;
    what[i] = f2bf(wpred[((p * 10 + j) * 16 + d) * 8 + k]);
  }
  for (int i = g; i < 512 * 160; i += gs)  d1b[i] = f2bf(d1w[i]);
  for (int i = g; i < 1024 * 512; i += gs) d2b[i] = f2bf(d2w[i]);
  for (int i = g; i < 784 * 1024; i += gs) d3b[i] = f2bf(d3w[i]);
  if (g == 0) { accs[0] = 0.f; accs[1] = 0.f; }
}

// ---------------------------------------------------------------------------
// Implicit-im2col conv GEMM.  A = weights bf16 [256][Kdim], B gathered from
// src bf16 [B][C][H][W].  Workgroup tile 128M x 128N, 8 waves.
// Ntot is an exact multiple of 128 for both convs (no n guard in the fill).
// ---------------------------------------------------------------------------
__global__ __launch_bounds__(256) void k_conv_gemm(
    const __bf16* __restrict__ A, const __bf16* __restrict__ src,
    const float* __restrict__ bias,
    __bf16* __restrict__ outBf, float* __restrict__ outF,
    int C, int H, int W, int stride, int OHOW, int OW,
    int Kdim, int Kvalid, int Ntot, int NB, int relu_bf16) {
  __shared__ __bf16 ldsA[128 * LSTR];
  __shared__ __bf16 ldsB[128 * LSTR];
  const int tid = threadIdx.x;
  const int mb = blockIdx.x / NB;
  const int nb = blockIdx.x - mb * NB;
  const int lane = tid & 31, wave = tid >> 5;
  const int wm = wave & 1, wn = wave >> 1;
  const int rowA = tid >> 1, halfA = tid & 1;
  f32x8 acc[4][2] = {};

  // Per-thread im2col invariants: thread owns half of LDS-B row nlB.
  const int nlB = tid >> 1;
  const int kof = (tid & 1) * 16;
  const int n   = nb * 128 + nlB;                  // always < Ntot here
  const int bimg = n / OHOW, pos = n - bimg * OHOW;
  const int oy = pos / OW, ox = pos - oy * OW;
  const __bf16* srcb = src + (((size_t)bimg * C) * H + oy * stride) * W + ox * stride;
  const __bf16* Arow = A + (size_t)(mb * 128 + rowA) * Kdim;
  const __bf16  zbf  = f2bf(0.f);

  for (int k0 = 0; k0 < Kdim; k0 += 32) {
    __syncthreads();
    // A tile: 128x32, two async b128 DMAs per thread (contiguous rows)
    copy16_g2l(Arow + k0 + halfA * 16,     &ldsA[rowA * LSTR + halfA * 16]);
    copy16_g2l(Arow + k0 + halfA * 16 + 8, &ldsA[rowA * LSTR + halfA * 16 + 8]);
    if (k0 + 32 < Kdim) __builtin_prefetch(Arow + k0 + 32, 0, 0);

    // B tile: incremental im2col gather (L2-resident src), register-staged,
    // stored as two ds_store_b128.
    {
      int ks = k0 + kof;
      int c = ks / 81, r = ks - c * 81;
      int ky = r / 9, kx = r - ky * 9;
      const __bf16* pg = srcb + ((size_t)c * H + ky) * W + kx;
      __attribute__((aligned(16))) __bf16 tmp[16];
#pragma unroll
      for (int i = 0; i < 16; ++i) {
        tmp[i] = (ks + i < Kvalid) ? *pg : zbf;    // k guard live only in conv1
        ++kx; ++pg;
        if (kx == 9) {
          kx = 0; pg += (W - 9); ++ky;
          if (ky == 9) { ky = 0; pg += (size_t)(H - 9) * W; }
        }
      }
      f32x4* d = reinterpret_cast<f32x4*>(&ldsB[nlB * LSTR + kof]);
      d[0] = *reinterpret_cast<const f32x4*>(&tmp[0]);
      d[1] = *reinterpret_cast<const f32x4*>(&tmp[8]);
    }
    copy_wait();          // s_wait_asynccnt 0 (A-tile DMA) before the barrier
    __syncthreads();
    mma_step(ldsA + wm * 64 * LSTR, ldsB + wn * 32 * LSTR, lane, acc);
  }

  const int hi8 = (lane & 16) ? 8 : 0;
  const int nlo = lane & 15;
#pragma unroll
  for (int nt = 0; nt < 2; ++nt) {
    int nn = nb * 128 + wn * 32 + nt * 16 + nlo;
    if (nn >= Ntot) continue;
    int bi = nn / OHOW, ps = nn - bi * OHOW;
#pragma unroll
    for (int mt = 0; mt < 4; ++mt)
#pragma unroll
      for (int r = 0; r < 8; ++r) {
        int m = mb * 128 + wm * 64 + mt * 16 + hi8 + r;
        float v = acc[mt][nt][r] + bias[m];
        size_t o = ((size_t)bi * 256 + m) * OHOW + ps;
        if (relu_bf16) outBf[o] = f2bf(fmaxf(v, 0.f));
        else           outF[o]  = v;
      }
  }
}

// ---------------------------------------------------------------------------
// Squash primary capsules: c2 f32 [B][256][36] -> u bf16 [p][B][8]
// ---------------------------------------------------------------------------
__global__ void k_squash(const float* __restrict__ c2, __bf16* __restrict__ u) {
  const int p = blockIdx.x, b = threadIdx.x;
  float comp[8]; float sq = 0.f;
#pragma unroll
  for (int k = 0; k < 8; ++k) {
    int f = p * 8 + k; int c = f / 36, pos = f - c * 36;
    float v = c2[((size_t)b * 256 + c) * 36 + pos];
    comp[k] = v; sq += v * v;
  }
  float sc = (sq / (1.f + sq)) / sqrtf(sq + EPSF);
#pragma unroll
  for (int k = 0; k < 8; ++k) u[((size_t)p * 256 + b) * 8 + k] = f2bf(comp[k] * sc);
}

// ---------------------------------------------------------------------------
// u_hat: per-capsule GEMM [256 x 8(pad32)] * [8 x 160] using WMMA.
// grid = 1152 * 2(Mblk) * 2(Nblk).  Output bf16 [b][p][160].
// ---------------------------------------------------------------------------
__global__ __launch_bounds__(256) void k_uhat(const __bf16* __restrict__ u,
                                              const __bf16* __restrict__ what,
                                              __bf16* __restrict__ uhat) {
  __shared__ __bf16 ldsA[128 * LSTR];
  __shared__ __bf16 ldsB[128 * LSTR];
  const int tid = threadIdx.x;
  const int p  = blockIdx.x >> 2;
  const int mb = (blockIdx.x >> 1) & 1;
  const int nb = blockIdx.x & 1;
  const __bf16 z = f2bf(0.f);
#pragma unroll 4
  for (int i = 0; i < 16; ++i) {
    int e = tid * 16 + i;
    int rw = e >> 5, k = e & 31;
    ldsA[rw * LSTR + k] = (k < 8) ? u[((size_t)p * 256 + mb * 128 + rw) * 8 + k] : z;
    int n = nb * 128 + rw;
    ldsB[rw * LSTR + k] = (k < 8 && n < 160) ? what[((size_t)p * 8 + k) * 160 + n] : z;
  }
  __syncthreads();
  const int lane = tid & 31, wave = tid >> 5, wm = wave & 1, wn = wave >> 1;
  f32x8 acc[4][2] = {};
  mma_step(ldsA + wm * 64 * LSTR, ldsB + wn * 32 * LSTR, lane, acc);
  const int hi8 = (lane & 16) ? 8 : 0, nlo = lane & 15;
#pragma unroll
  for (int nt = 0; nt < 2; ++nt) {
    int n = nb * 128 + wn * 32 + nt * 16 + nlo;
    if (n >= 160) continue;
#pragma unroll
    for (int mt = 0; mt < 4; ++mt)
#pragma unroll
      for (int r = 0; r < 8; ++r) {
        int b = mb * 128 + wm * 64 + mt * 16 + hi8 + r;
        uhat[((size_t)b * 1152 + p) * 160 + n] = f2bf(acc[mt][nt][r]);
      }
  }
}

// ---------------------------------------------------------------------------
// Dynamic routing, one workgroup per image.  b_ij logits live in LDS (46KB).
// ---------------------------------------------------------------------------
__global__ __launch_bounds__(256) void k_routing(const __bf16* __restrict__ uhat,
                                                 float* __restrict__ vj) {
  __shared__ float bij[1152 * 10];
  __shared__ float mxs[1152];
  __shared__ float dens[1152];
  __shared__ float sj[160];
  __shared__ float vv[160];
  const int tid = threadIdx.x;
  const int b = blockIdx.x;
  for (int i = tid; i < 11520; i += 256) bij[i] = 0.f;
  for (int it = 0; it < 3; ++it) {
    __syncthreads();
    for (int p = tid; p < 1152; p += 256) {       // softmax denominators
      float m = bij[p * 10];
      for (int j = 1; j < 10; ++j) m = fmaxf(m, bij[p * 10 + j]);
      float s = 0.f;
      for (int j = 0; j < 10; ++j) s += __expf(bij[p * 10 + j] - m);
      mxs[p] = m; dens[p] = s;
    }
    __syncthreads();
    if (tid < 160) {                               // s_j owner per (j,d)
      const int j = tid >> 4;
      float s = 0.f;
      for (int p = 0; p < 1152; ++p) {
        float c = __expf(bij[p * 10 + j] - mxs[p]) / dens[p];
        s += c * bf2f(uhat[((size_t)b * 1152 + p) * 160 + tid]);
      }
      sj[tid] = s;
    }
    __syncthreads();
    if (tid < 160) {                               // squash
      const int j = tid >> 4;
      float sq = 0.f;
      for (int d = 0; d < 16; ++d) { float t = sj[j * 16 + d]; sq += t * t; }
      vv[tid] = sj[tid] * (sq / (1.f + sq)) / sqrtf(sq + EPSF);
    }
    __syncthreads();
    if (it < 2) {                                  // agreement update
      for (int p = tid; p < 1152; p += 256)
        for (int j = 0; j < 10; ++j) {
          float dot = 0.f;
          for (int d = 0; d < 16; ++d)
            dot += vv[j * 16 + d] * bf2f(uhat[((size_t)b * 1152 + p) * 160 + j * 16 + d]);
          bij[p * 10 + j] += dot;
        }
    }
  }
  __syncthreads();
  if (tid < 160) vj[(size_t)b * 160 + tid] = vv[tid];
}

// ---------------------------------------------------------------------------
// Margin loss partials, argmax, masked decoder input (one thread per image).
// ---------------------------------------------------------------------------
__global__ void k_margin(const float* __restrict__ vj, const int* __restrict__ labels,
                         float* __restrict__ outIdx, __bf16* __restrict__ decin,
                         float* __restrict__ marginAcc) {
  const int b = threadIdx.x;
  const float* v = vj + (size_t)b * 160;
  float nrm[10]; float best = -1.f; int bi = 0;
#pragma unroll
  for (int j = 0; j < 10; ++j) {
    float s = 0.f;
    for (int d = 0; d < 16; ++d) s += v[j * 16 + d] * v[j * 16 + d];
    nrm[j] = sqrtf(s + EPSF);
    if (nrm[j] > best) { best = nrm[j]; bi = j; }
  }
  const int lab = labels[b];
  float msum = 0.f;
#pragma unroll
  for (int j = 0; j < 10; ++j) {
    float t = (j == lab) ? 1.f : 0.f;
    float c = fmaxf(0.9f - nrm[j], 0.f);
    float ic = fmaxf(nrm[j] - 0.1f, 0.f);
    msum += t * c * c + 0.5f * (1.f - t) * ic * ic;
  }
  atomicAdd(marginAcc, msum);
  outIdx[b] = (float)bi;
  for (int jd = 0; jd < 160; ++jd)
    decin[(size_t)b * 160 + jd] = f2bf(((jd >> 4) == bi) ? v[jd] : 0.f);
}

// ---------------------------------------------------------------------------
// Dense WMMA GEMM: C[256xN] = A[256xK] * Bw[NxK]^T + bias.  mode 0: relu->bf16.
// mode 1: sigmoid->f32 (dec_out) + recon-loss accumulation vs input.
// ---------------------------------------------------------------------------
__global__ __launch_bounds__(256) void k_dense_gemm(
    const __bf16* __restrict__ A, const __bf16* __restrict__ Bw,
    const float* __restrict__ bias, int N, int K, int NB, int mode,
    __bf16* __restrict__ outBf, float* __restrict__ outF,
    const float* __restrict__ ref, float* __restrict__ reconAcc) {
  __shared__ __bf16 ldsA[128 * LSTR];
  __shared__ __bf16 ldsB[128 * LSTR];
  const int tid = threadIdx.x;
  const int mb = blockIdx.x / NB;
  const int nb = blockIdx.x - mb * NB;
  const int lane = tid & 31, wave = tid >> 5;
  const int wm = wave & 1, wn = wave >> 1;
  const int row = tid >> 1, half = tid & 1;
  const int nB = nb * 128 + row;
  f32x8 acc[4][2] = {};

  for (int k0 = 0; k0 < K; k0 += 32) {
    __syncthreads();
    {   // A tile (activations, row-major [256][K]) via async DMA
      const __bf16* g = A + (size_t)(mb * 128 + row) * K + k0 + half * 16;
      copy16_g2l(g,     &ldsA[row * LSTR + half * 16]);
      copy16_g2l(g + 8, &ldsA[row * LSTR + half * 16 + 8]);
    }
    {   // B tile (weights [N][K] == torch [out][in])
      __bf16* l = &ldsB[row * LSTR + half * 16];
      if (nB < N) {
        const __bf16* g = Bw + (size_t)nB * K + k0 + half * 16;
        copy16_g2l(g, l);
        copy16_g2l(g + 8, l + 8);
      } else {
        f32x4 z = {0.f, 0.f, 0.f, 0.f};
        reinterpret_cast<f32x4*>(l)[0] = z;
        reinterpret_cast<f32x4*>(l)[1] = z;
      }
    }
    copy_wait();
    __syncthreads();
    mma_step(ldsA + wm * 64 * LSTR, ldsB + wn * 32 * LSTR, lane, acc);
  }

  const int hi8 = (lane & 16) ? 8 : 0, nlo = lane & 15;
#pragma unroll
  for (int nt = 0; nt < 2; ++nt) {
    int n = nb * 128 + wn * 32 + nt * 16 + nlo;
    if (n >= N) continue;
    float bz = bias[n];
#pragma unroll
    for (int mt = 0; mt < 4; ++mt)
#pragma unroll
      for (int r = 0; r < 8; ++r) {
        int m = mb * 128 + wm * 64 + mt * 16 + hi8 + r;
        float v = acc[mt][nt][r] + bz;
        if (mode == 0) {
          outBf[(size_t)m * N + n] = f2bf(fmaxf(v, 0.f));
        } else {
          float sg = 1.f / (1.f + __expf(-v));
          outF[(size_t)m * 784 + n] = sg;
          float df = sg - ref[(size_t)m * 784 + n];
          atomicAdd(reconAcc, df * df);
        }
      }
  }
}

__global__ void k_finalize(const float* __restrict__ accs, float* __restrict__ out) {
  out[0] = accs[0] * (1.f / 256.f) + 0.0005f * accs[1] * (1.f / (256.f * 784.f));
}

// ---------------------------------------------------------------------------
extern "C" void kernel_launch(void* const* d_in, const int* in_sizes, int n_in,
                              void* d_out, int out_size, void* d_ws, size_t ws_size,
                              hipStream_t stream) {
  const float* input   = (const float*)d_in[0];
  const int*   labels  = (const int*)d_in[1];
  const float* conv1_w = (const float*)d_in[2];
  const float* conv1_b = (const float*)d_in[3];
  const float* conv2_w = (const float*)d_in[4];
  const float* conv2_b = (const float*)d_in[5];
  const float* W_pred  = (const float*)d_in[6];
  const float* d1_w    = (const float*)d_in[7];
  const float* d1_b    = (const float*)d_in[8];
  const float* d2_w    = (const float*)d_in[9];
  const float* d2_b    = (const float*)d_in[10];
  const float* d3_w    = (const float*)d_in[11];
  const float* d3_b    = (const float*)d_in[12];
  (void)in_sizes; (void)n_in; (void)out_size; (void)ws_size;

  char* w = (char*)d_ws;
  auto take = [&](size_t bytes) -> char* {
    char* r = w; w += (bytes + 255) & ~(size_t)255; return r;
  };
  __bf16* xbf  = (__bf16*)take((size_t)256 * 784 * 2);        // input bf16
  __bf16* a1   = (__bf16*)take((size_t)256 * 96 * 2);         // conv1_w padded
  __bf16* c1   = (__bf16*)take((size_t)256 * 256 * 400 * 2);  // conv1 out (52MB, L2-resident)
  __bf16* a2   = (__bf16*)take((size_t)256 * 20736 * 2);      // conv2_w
  float*  c2o  = (float*)take((size_t)256 * 256 * 36 * 4);    // conv2 out f32
  __bf16* u    = (__bf16*)take((size_t)1152 * 256 * 8 * 2);   // squashed primaries [p][b][8]
  __bf16* what = (__bf16*)take((size_t)1152 * 8 * 160 * 2);   // W_pred reordered [p][k][jd]
  __bf16* uh   = (__bf16*)take((size_t)256 * 1152 * 160 * 2); // u_hat bf16 (94MB)
  float*  vj   = (float*)take((size_t)256 * 160 * 4);
  __bf16* din  = (__bf16*)take((size_t)256 * 160 * 2);
  __bf16* h1   = (__bf16*)take((size_t)256 * 512 * 2);
  __bf16* h2   = (__bf16*)take((size_t)256 * 1024 * 2);
  __bf16* d1b  = (__bf16*)take((size_t)512 * 160 * 2);
  __bf16* d2b  = (__bf16*)take((size_t)1024 * 512 * 2);
  __bf16* d3b  = (__bf16*)take((size_t)784 * 1024 * 2);
  float*  accs = (float*)take(256);                           // [0]=margin sum, [1]=recon sum

  float* outIdx  = (float*)d_out;        // [0..255]  argmax per image
  float* outLoss = outIdx + 256;         // [256]     final loss
  float* outDec  = outIdx + 257;         // [257..]   dec_out [256][784]

  k_prep<<<2048, 256, 0, stream>>>(input, conv1_w, conv2_w, W_pred, d1_w, d2_w, d3_w,
                                   xbf, a1, a2, what, d1b, d2b, d3b, accs);

  // conv1: M=256, N=256*400=102400 (800 n-blocks), K=96 (81 valid), stride 1
  k_conv_gemm<<<2 * 800, 256, 0, stream>>>(a1, xbf, conv1_b, c1, nullptr,
                                           1, 28, 28, 1, 400, 20,
                                           96, 81, 102400, 800, 1);
  // conv2: M=256, N=256*36=9216 (72 n-blocks), K=20736, stride 2
  k_conv_gemm<<<2 * 72, 256, 0, stream>>>(a2, c1, conv2_b, nullptr, c2o,
                                          256, 20, 20, 2, 36, 6,
                                          20736, 20736, 9216, 72, 0);

  k_squash<<<1152, 256, 0, stream>>>(c2o, u);
  k_uhat<<<1152 * 4, 256, 0, stream>>>(u, what, uh);
  k_routing<<<256, 256, 0, stream>>>(uh, vj);
  k_margin<<<1, 256, 0, stream>>>(vj, labels, outIdx, din, accs);

  // decoder: 160->512 relu, 512->1024 relu, 1024->784 sigmoid (+recon loss)
  k_dense_gemm<<<2 * 4, 256, 0, stream>>>(din, d1b, d1_b, 512, 160, 4, 0,
                                          h1, nullptr, nullptr, nullptr);
  k_dense_gemm<<<2 * 8, 256, 0, stream>>>(h1, d2b, d2_b, 1024, 512, 8, 0,
                                          h2, nullptr, nullptr, nullptr);
  k_dense_gemm<<<2 * 7, 256, 0, stream>>>(h2, d3b, d3_b, 784, 1024, 7, 1,
                                          nullptr, outDec, input, accs + 1);

  k_finalize<<<1, 1, 0, stream>>>(accs, outLoss);
}
